// xLSTM_54022098649617
// MI455X (gfx1250) — compile-verified, matrix operand order checked by reading, not run on hardware
//
#include <hip/hip_runtime.h>
#include <math.h>

#define B_    512
#define D_    768
#define H_    8
#define DH_   96
#define C_    768       // H*DH
#define PROJ_ 1024
#define PFD_  1024
#define KER_  4
#define OUT_  10
#define EPSF  1e-5f

typedef __attribute__((ext_vector_type(16))) __bf16 v16bf;
typedef __attribute__((ext_vector_type(4)))  __bf16 v4bf;
typedef __attribute__((ext_vector_type(8)))  float  v8f;

__device__ __forceinline__ __bf16 f2bf(float f) {
  union { float f; unsigned u; } c; c.f = f;
  unsigned r = (c.u + 0x7FFFu + ((c.u >> 16) & 1u)) >> 16;
  union { unsigned short s; __bf16 b; } o; o.s = (unsigned short)r;
  return o.b;
}

// ---- gfx1250 async global->LDS (ASYNCcnt) ---------------------------------
__device__ __forceinline__ void async_ld16(unsigned lds_addr, const __bf16* gaddr) {
  asm volatile("global_load_async_to_lds_b128 %0, %1, off"
               :: "v"(lds_addr), "v"(gaddr) : "memory");
}
__device__ __forceinline__ void wait_async0() {
  asm volatile("s_wait_asynccnt 0x0" ::: "memory");
}

// ---------------------------------------------------------------------------
// WMMA GEMM, bf16 inputs: Y[M,N] = epi( Xb[M,K] @ Wb[N,K]^T )
// epi: v = scale*(acc + bias[n]); v += addsrc[m,n]; if(flags&1) sigmoid;
//      store f32 Y, optionally bf16 Ybf.
// Requires: M%128==0, N%64==0, K%32==0.
// 256 thr = 8 waves; block tile 128x64; wave tile 32x32; double-buffered LDS
// filled by GLOBAL_LOAD_ASYNC_TO_LDS_B128 overlapped with WMMA.
// ---------------------------------------------------------------------------
#define TM 128
#define TN 64
#define TK 32
#define LDA 40   // bf16 row stride in LDS (32 + 8 pad)

__global__ __launch_bounds__(256)
void gemm_bf16_async(const __bf16* __restrict__ Xb, const __bf16* __restrict__ Wb,
                     const float* __restrict__ bias, const float* __restrict__ addsrc,
                     float* __restrict__ Y, __bf16* __restrict__ Ybf,
                     int M, int N, int K, float scale, int flags) {
  __shared__ __bf16 As[2][TM * LDA];
  __shared__ __bf16 Bs[2][TN * LDA];
  const int tid  = threadIdx.x;
  const int lane = tid & 31;
  const int wv   = tid >> 5;
  const int wm   = wv & 3;
  const int wn   = wv >> 2;
  const int mBlock = blockIdx.y * TM;
  const int nBlock = blockIdx.x * TN;

  v8f zero = {0.f,0.f,0.f,0.f,0.f,0.f,0.f,0.f};
  v8f acc[2][2];
  acc[0][0]=zero; acc[0][1]=zero; acc[1][0]=zero; acc[1][1]=zero;

  // async stage of one 128x32 A tile (512 x 16B chunks) + 64x32 B tile (256)
  auto issue = [&](int k0, int sel) {
    unsigned ab = (unsigned)(size_t)(&As[sel][0]);
    unsigned bb = (unsigned)(size_t)(&Bs[sel][0]);
#pragma unroll
    for (int i = 0; i < 2; ++i) {
      int ch = tid + i*256;
      int row = ch >> 2, q = ch & 3;
      async_ld16(ab + (unsigned)(row*LDA*2 + q*16),
                 Xb + (size_t)(mBlock + row)*K + k0 + q*8);
    }
    {
      int row = tid >> 2, q = tid & 3;
      async_ld16(bb + (unsigned)(row*LDA*2 + q*16),
                 Wb + (size_t)(nBlock + row)*K + k0 + q*8);
    }
  };

  const int nt = K / TK;
  issue(0, 0);
  wait_async0();
  __syncthreads();

  const int fr = lane & 15;
  const int kb = (lane >> 4) << 3;

  for (int t = 0; t < nt; ++t) {
    const int sel = t & 1;
    if (t + 1 < nt) issue((t+1)*TK, sel ^ 1);   // overlap copy with compute

    v16bf afr[2], bfr[2];
#pragma unroll
    for (int s = 0; s < 2; ++s) {
      const __bf16* pa = &As[sel][(wm*32 + s*16 + fr)*LDA + kb];
      const __bf16* pb = &Bs[sel][(wn*32 + s*16 + fr)*LDA + kb];
#pragma unroll
      for (int e = 0; e < 8; ++e) {
        afr[s][e] = pa[e];  afr[s][8+e] = pa[16+e];
        bfr[s][e] = pb[e];  bfr[s][8+e] = pb[16+e];
      }
    }
#pragma unroll
    for (int i = 0; i < 2; ++i)
#pragma unroll
      for (int j = 0; j < 2; ++j)
        acc[i][j] = __builtin_amdgcn_wmma_f32_16x16x32_bf16(
            false, afr[i], false, bfr[j], (short)0, acc[i][j], false, false);

    wait_async0();      // next tile landed (overlapped with WMMAs above)
    __syncthreads();
  }

  const int ln = lane & 15;
  const int hi = lane >> 4;
#pragma unroll
  for (int i = 0; i < 2; ++i) {
#pragma unroll
    for (int j = 0; j < 2; ++j) {
      int gc = nBlock + wn*32 + j*16 + ln;
      float bv = bias ? bias[gc] : 0.f;
#pragma unroll
      for (int r = 0; r < 8; ++r) {
        int gr = mBlock + wm*32 + i*16 + hi*8 + r;
        float v = (acc[i][j][r] + bv) * scale;
        if (addsrc) v += addsrc[(size_t)gr*N + gc];
        if (flags & 1) v = 1.f/(1.f + expf(-v));
        Y[(size_t)gr*N + gc] = v;
        if (Ybf) Ybf[(size_t)gr*N + gc] = f2bf(v);
      }
    }
  }
}

// ---------------------------------------------------------------------------
__global__ void f2bf_kernel(const float* __restrict__ x, __bf16* __restrict__ y, int n4) {
  int t = blockIdx.x*blockDim.x + threadIdx.x;
  if (t >= n4) return;
  const float4 f = ((const float4*)x)[t];
  v4bf v; v[0]=f2bf(f.x); v[1]=f2bf(f.y); v[2]=f2bf(f.z); v[3]=f2bf(f.w);
  ((v4bf*)y)[t] = v;
}

__global__ void small_gemm_bf(const __bf16* __restrict__ X, const __bf16* __restrict__ W,
                              const float* __restrict__ bias, float* __restrict__ Y,
                              int M, int N, int K) {
  int t = blockIdx.x*blockDim.x + threadIdx.x;
  if (t >= M*N) return;
  int row = t / N, col = t % N;
  const __bf16* xr = X + (size_t)row*K;
  const __bf16* wr = W + (size_t)col*K;
  float acc = bias ? bias[col] : 0.f;
  for (int k = 0; k < K; ++k) acc += (float)xr[k] * (float)wr[k];
  Y[t] = acc;
}

__global__ __launch_bounds__(256)
void ln_kernel(const float* __restrict__ x, const float* __restrict__ g,
               const float* __restrict__ b, __bf16* __restrict__ y, int D) {
  __shared__ float s1[256], s2[256];
  int row = blockIdx.x, tid = threadIdx.x;
  const float* xr = x + (size_t)row*D;
  float a = 0.f, q = 0.f;
  for (int j = tid; j < D; j += 256) { float v = xr[j]; a += v; q += v*v; }
  s1[tid] = a; s2[tid] = q; __syncthreads();
  for (int s = 128; s > 0; s >>= 1) {
    if (tid < s) { s1[tid] += s1[tid+s]; s2[tid] += s2[tid+s]; }
    __syncthreads();
  }
  float mu = s1[0] / D;
  float var = s2[0] / D - mu*mu;
  float rs = rsqrtf(var + EPSF);
  __bf16* yr = y + (size_t)row*D;
  for (int j = tid; j < D; j += 256) yr[j] = f2bf((xr[j]-mu)*rs*g[j] + b[j]);
}

// group norm (bf16 out); optional skip add and silu(r) gating
__global__ __launch_bounds__(256)
void gn_kernel(const float* __restrict__ x, const float* __restrict__ g,
               const float* __restrict__ b, const float* __restrict__ skip,
               const float* __restrict__ rmul, __bf16* __restrict__ y,
               int Cc, int groups) {
  __shared__ float gs[8], gq[8];
  int row = blockIdx.x, tid = threadIdx.x;
  int gsz = Cc / groups;
  if (tid < groups) { gs[tid] = 0.f; gq[tid] = 0.f; }
  __syncthreads();
  const float* xr = x + (size_t)row*Cc;
  for (int j = tid; j < Cc; j += 256) {
    float v = xr[j];
    atomicAdd(&gs[j/gsz], v);        // ds_add_f32
    atomicAdd(&gq[j/gsz], v*v);
  }
  __syncthreads();
  __bf16* yr = y + (size_t)row*Cc;
  for (int j = tid; j < Cc; j += 256) {
    int gi = j / gsz;
    float mu = gs[gi] / gsz;
    float var = gq[gi] / gsz - mu*mu;
    float v = (xr[j]-mu)*rsqrtf(var+EPSF)*g[j] + b[j];
    if (skip) v += skip[(size_t)row*Cc + j];
    if (rmul) { float r = rmul[(size_t)row*Cc + j]; v *= r/(1.f+expf(-r)); }
    yr[j] = f2bf(v);
  }
}

__global__ void slstm_point(const float* __restrict__ gi, const float* __restrict__ gf,
                            const float* __restrict__ gz, const float* __restrict__ go,
                            const float* __restrict__ c0, const float* __restrict__ n0,
                            const float* __restrict__ m0,
                            float* __restrict__ c1, float* __restrict__ n1,
                            float* __restrict__ h1, float* __restrict__ m1, int total) {
  int t = blockIdx.x*blockDim.x + threadIdx.x;
  if (t >= total) return;
  float i = gi[t], f = gf[t], z = gz[t], o = go[t];
  float mo = m0[t];
  float mt = fmaxf(f + mo, i);
  float it = expf(i - mt);
  float ft = expf(f - mt + mo);
  float ct = ft*c0[t] + it*tanhf(z);
  float nt = ft*n0[t] + it;
  float ht = (1.f/(1.f+expf(-o))) * (ct/nt);
  c1[t]=ct; n1[t]=nt; h1[t]=ht; m1[t]=mt;
}

__global__ void gelu_comb(const float* __restrict__ up, __bf16* __restrict__ y, int total) {
  int t = blockIdx.x*blockDim.x + threadIdx.x;
  if (t >= total) return;
  int row = t / PROJ_, j = t % PROJ_;
  const float* ur = up + (size_t)row*(2*PROJ_);
  float x1 = ur[j], x2 = ur[PROJ_ + j];
  float ge = 0.5f*x2*(1.f + erff(x2*0.7071067811865475f));
  y[t] = f2bf(x1 + ge);
}

__global__ void conv_silu(const float* __restrict__ x, const float* __restrict__ w,
                          const float* __restrict__ bias, __bf16* __restrict__ y,
                          int L, int total) {
  int t = blockIdx.x*blockDim.x + threadIdx.x;
  if (t >= total) return;
  int row = t / L, j = t % L;
  const float* xr = x + (size_t)row*L;
  float acc = bias[0];
#pragma unroll
  for (int kk = 0; kk < KER_; ++kk) {
    int idx = j - (KER_-1) + kk;
    if (idx >= 0) acc += w[kk]*xr[idx];
  }
  y[t] = f2bf(acc/(1.f + expf(-acc)));
}

// one workgroup per (b,h); 96 active threads each own a row d of c[96][96]
__global__ __launch_bounds__(128)
void mlstm_state(const float* __restrict__ c0, const float* __restrict__ n0,
                 const float* __restrict__ m0,
                 const float* __restrict__ qv, const float* __restrict__ kv,
                 const float* __restrict__ vv,
                 const float* __restrict__ ipre, const float* __restrict__ fpre,
                 const float* __restrict__ osig,
                 float* __restrict__ c1, float* __restrict__ n1,
                 float* __restrict__ m1, float* __restrict__ ht) {
  __shared__ float sq[DH_], sk[DH_], sv[DH_], part[DH_], sden;
  int blk = blockIdx.x;
  int b = blk / H_, h = blk % H_;
  int tid = threadIdx.x;
  float ip = ipre[b*H_ + h];
  float fp = fpre[b*H_ + h];
  float mo = m0[b*H_ + h];
  float mt = fmaxf(fp + mo, ip);
  float it = expf(ip - mt);
  float ft = expf(fp - mt + mo);
  if (tid == 0) m1[b*H_ + h] = mt;
  size_t cbase = (size_t)(b*H_ + h) * DH_ * DH_;
  size_t vbase = (size_t)b*C_ + h*DH_;
  if (tid < DH_) {
    sq[tid] = qv[vbase + tid];
    sk[tid] = kv[vbase + tid];
    sv[tid] = vv[vbase + tid];
  }
  __syncthreads();
  float num = 0.f;
  if (tid < DH_) {
    const float* cr = c0 + cbase + (size_t)tid*DH_;
    float*       co = c1 + cbase + (size_t)tid*DH_;
    float vd = sv[tid];
    for (int p = 0; p < DH_; ++p) {
      float cn = ft*cr[p] + it*vd*sk[p];
      co[p] = cn;
      num += cn*sq[p];
    }
    size_t nidx = (size_t)(b*H_ + h)*DH_ + tid;
    float nt = ft*n0[nidx] + it*sk[tid];
    n1[nidx] = nt;
    part[tid] = nt*sq[tid];
  }
  __syncthreads();
  if (tid == 0) {
    float s = 0.f;
    for (int d = 0; d < DH_; ++d) s += part[d];
    sden = fmaxf(s, 1.0f);
  }
  __syncthreads();
  if (tid < DH_) ht[vbase + tid] = osig[vbase + tid] * num / sden;
}

// ---------------------------------------------------------------------------
extern "C" void kernel_launch(void* const* d_in, const int* in_sizes, int n_in,
                              void* d_out, int out_size, void* d_ws, size_t ws_size,
                              hipStream_t stream) {
  (void)in_sizes; (void)n_in; (void)out_size; (void)ws_size;
  const int B = B_, D = D_, C = C_, H = H_, DH = DH_;
  const int BC = B*C;

  auto fin  = [&](int i){ return (const float*)d_in[i]; };
  // recursive dict-insertion-order flattening of setup_inputs():
  // 0: seq | 1..24/25..48: slstm[0/1] (24 each) | 49..73/74..98: mlstm[0/1] (25 each)
  // 99: out_w 100: out_b | 101..108: s_hid (c,n,h,m)x2 | 109..114: m_hid (c,n,m)x2
  auto sl   = [&](int li, int o){ return fin(1  + li*24 + o); };
  auto ml   = [&](int li, int o){ return fin(49 + li*25 + o); };
  const float* out_w = fin(99);
  const float* out_b = fin(100);
  auto shid = [&](int li, int o){ return fin(101 + li*4 + o); };
  auto mhid = [&](int li, int o){ return fin(109 + li*3 + o); };

  // d_out: logits, then (c,n,h,m) per sLSTM layer, then (C,n,m) per mLSTM layer
  float* out = (float*)d_out;
  float* logits = out;
  size_t off = (size_t)B*OUT_;
  float *sOc[2], *sOn[2], *sOh[2], *sOm[2];
  for (int li = 0; li < 2; ++li) {
    sOc[li]=out+off; off+=BC; sOn[li]=out+off; off+=BC;
    sOh[li]=out+off; off+=BC; sOm[li]=out+off; off+=BC;
  }
  float *mOc[2], *mOn[2], *mOm[2];
  for (int li = 0; li < 2; ++li) {
    mOc[li]=out+off; off+=(size_t)B*H*DH*DH;
    mOn[li]=out+off; off+=(size_t)B*H*DH;
    mOm[li]=out+off; off+=(size_t)B*H;
  }

  // ---- workspace: f32 region then bf16 region ----
  float* ws = (float*)d_ws; size_t o2 = 0;
  auto alloc = [&](size_t n){ float* p = ws + o2; o2 += n; return p; };
  float* bufA = alloc((size_t)B*D);
  float* bufB = alloc((size_t)B*D);
  float* g_i  = alloc((size_t)BC);
  float* g_f  = alloc((size_t)BC);
  float* g_z  = alloc((size_t)BC);
  float* g_o  = alloc((size_t)BC);
  float* up   = alloc((size_t)B*2*PROJ_);
  float* x_t  = alloc((size_t)B*PFD_);
  float* r_t  = alloc((size_t)BC);
  float* qb   = alloc((size_t)BC);
  float* kbuf = alloc((size_t)BC);
  float* vbuf = alloc((size_t)BC);
  float* prei = alloc((size_t)B*H);
  float* pref = alloc((size_t)B*H);
  float* osg  = alloc((size_t)BC);
  float* htm  = alloc((size_t)BC);
  float* skp  = alloc((size_t)BC);

  __bf16* hb = (__bf16*)(ws + o2); size_t o3 = 0;
  auto halloc = [&](size_t n){ __bf16* p = hb + o3; o3 += n; return p; };
  __bf16* wreg   = halloc(7077888);           // per-sub-block weight region (reused)
  __bf16* outwbf = halloc((size_t)OUT_*C);
  __bf16* xn_bf  = halloc((size_t)B*D);
  __bf16* hp_bf  = halloc((size_t)BC);
  __bf16* gno_bf = halloc((size_t)BC);
  __bf16* h1_bf  = halloc((size_t)B*PROJ_);
  __bf16* xt_bf  = halloc((size_t)B*PFD_);
  __bf16* xc_bf  = halloc((size_t)B*PFD_);
  __bf16* cmb_bf = halloc((size_t)BC);
  __bf16* res_bf = halloc((size_t)B*D);

  auto cw = [&](const float* src, __bf16* dst, size_t n){
    int n4 = (int)(n/4);
    f2bf_kernel<<<(n4+255)/256, 256, 0, stream>>>(src, dst, n4);
  };
  auto gemm = [&](const __bf16* X, const __bf16* W, const float* bias,
                  const float* add, float* Y, __bf16* Ybf,
                  int M, int N, int K, float scale, int flags){
    dim3 grid(N/TN, M/TM);
    gemm_bf16_async<<<grid, 256, 0, stream>>>(X, W, bias, add, Y, Ybf,
                                              M, N, K, scale, flags);
  };

  // sLSTM weight slots in wreg
  const size_t CD = (size_t)C*D, CC = (size_t)C*C;
  __bf16 *wZ=wreg, *wI=wreg+CD, *wO=wreg+2*CD, *wF=wreg+3*CD;
  __bf16 *wRz=wreg+4*CD, *wRi=wreg+4*CD+CC, *wRo=wreg+4*CD+2*CC, *wRf=wreg+4*CD+3*CC;
  __bf16 *wUp=wreg+4*CD+4*CC, *wDn=wUp+(size_t)2*PROJ_*C;
  // mLSTM weight slots (region reused after sLSTM GEMMs complete in-stream)
  const size_t PD=(size_t)PFD_*D, CP=(size_t)C*PFD_;
  __bf16 *mUpl=wreg, *mUpr=wreg+PD, *mDn=mUpr+CC, *mSkip=mDn+CC, *mWo=mSkip+CP,
         *mWq=mWo+CP, *mWk=mWq+CP, *mWv=mWk+CP, *mWi=mWv+CP, *mWf=mWi+(size_t)H*PFD_;

  cw(out_w, outwbf, (size_t)OUT_*C);

  const float kscale = 0.102062072615966f; // 1/sqrt(96)
  const float* cur = (const float*)d_in[0];

  for (int li = 0; li < 2; ++li) {
    // ======== sLSTM ========
    cw(sl(li,4),  wZ,  CD);  cw(sl(li,6),  wI,  CD);
    cw(sl(li,8),  wO,  CD);  cw(sl(li,10), wF,  CD);
    cw(sl(li,12), wRz, CC);  cw(sl(li,14), wRi, CC);
    cw(sl(li,16), wRo, CC);  cw(sl(li,18), wRf, CC);
    cw(sl(li,20), wUp, (size_t)2*PROJ_*C);
    cw(sl(li,22), wDn, (size_t)D*PROJ_);
    cw(shid(li,2), hp_bf, (size_t)BC);

    ln_kernel<<<B, 256, 0, stream>>>(cur, sl(li,0), sl(li,1), xn_bf, D);
    gemm(xn_bf, wI,  sl(li,7),  nullptr, g_i, nullptr, B, C, D, 1.f, 0);
    gemm(hp_bf, wRi, sl(li,15), g_i,     g_i, nullptr, B, C, C, 1.f, 0);
    gemm(xn_bf, wF,  sl(li,11), nullptr, g_f, nullptr, B, C, D, 1.f, 0);
    gemm(hp_bf, wRf, sl(li,19), g_f,     g_f, nullptr, B, C, C, 1.f, 0);
    gemm(xn_bf, wZ,  sl(li,5),  nullptr, g_z, nullptr, B, C, D, 1.f, 0);
    gemm(hp_bf, wRz, sl(li,13), g_z,     g_z, nullptr, B, C, C, 1.f, 0);
    gemm(xn_bf, wO,  sl(li,9),  nullptr, g_o, nullptr, B, C, D, 1.f, 0);
    gemm(hp_bf, wRo, sl(li,17), g_o,     g_o, nullptr, B, C, C, 1.f, 0);
    slstm_point<<<(BC+255)/256, 256, 0, stream>>>(g_i, g_f, g_z, g_o,
        shid(li,0), shid(li,1), shid(li,3),
        sOc[li], sOn[li], sOh[li], sOm[li], BC);
    gn_kernel<<<B, 256, 0, stream>>>(sOh[li], sl(li,2), sl(li,3),
                                     nullptr, nullptr, gno_bf, C, H);
    gemm(gno_bf, wUp, sl(li,21), nullptr, up, nullptr, B, 2*PROJ_, C, 1.f, 0);
    gelu_comb<<<(B*PROJ_+255)/256, 256, 0, stream>>>(up, h1_bf, B*PROJ_);
    gemm(h1_bf, wDn, sl(li,23), cur, bufA, nullptr, B, D, PROJ_, 1.f, 0);

    // ======== mLSTM ========
    cw(ml(li,4),  mUpl, PD);  cw(ml(li,6),  mUpr, CC);
    cw(ml(li,8),  mDn,  CC);  cw(ml(li,12), mSkip, CP);
    cw(ml(li,17), mWo,  CP);  cw(ml(li,19), mWq,  CP);
    cw(ml(li,21), mWk,  CP);  cw(ml(li,23), mWv,  CP);
    cw(ml(li,13), mWi,  (size_t)H*PFD_);
    cw(ml(li,15), mWf,  (size_t)H*PFD_);

    ln_kernel<<<B, 256, 0, stream>>>(bufA, ml(li,0), ml(li,1), xn_bf, D);
    gemm(xn_bf, mUpl, ml(li,5), nullptr, x_t, xt_bf, B, PFD_, D, 1.f, 0);
    gemm(xn_bf, mUpr, ml(li,7), nullptr, r_t, nullptr, B, C, D, 1.f, 0);
    conv_silu<<<(B*PFD_+255)/256, 256, 0, stream>>>(x_t, ml(li,10), ml(li,11),
                                                    xc_bf, PFD_, B*PFD_);
    gemm(xc_bf, mWq, ml(li,20), nullptr, qb,   nullptr, B, C, PFD_, 1.f,    0);
    gemm(xc_bf, mWk, ml(li,22), nullptr, kbuf, nullptr, B, C, PFD_, kscale, 0);
    gemm(xt_bf, mWv, ml(li,24), nullptr, vbuf, nullptr, B, C, PFD_, 1.f,    0);
    small_gemm_bf<<<(B*H+255)/256, 256, 0, stream>>>(xc_bf, mWi, ml(li,14), prei, B, H, PFD_);
    small_gemm_bf<<<(B*H+255)/256, 256, 0, stream>>>(xc_bf, mWf, ml(li,16), pref, B, H, PFD_);
    gemm(xt_bf, mWo, ml(li,18), nullptr, osg, nullptr, B, C, PFD_, 1.f, 1);
    mlstm_state<<<B*H, 128, 0, stream>>>(mhid(li,0), mhid(li,1), mhid(li,2),
        qb, kbuf, vbuf, prei, pref, osg,
        mOc[li], mOn[li], mOm[li], htm);
    gemm(xc_bf, mSkip, nullptr, nullptr, skp, nullptr, B, C, PFD_, 1.f, 0);
    gn_kernel<<<B, 256, 0, stream>>>(htm, ml(li,2), ml(li,3), skp, r_t, cmb_bf, C, H);
    gemm(cmb_bf, mDn, ml(li,9), bufA, bufB, res_bf, B, D, C, 1.f, 0);
    cur = bufB;
  }

  small_gemm_bf<<<(B_*OUT_+255)/256, 256, 0, stream>>>(res_bf, outwbf, out_b, logits,
                                                       B, OUT_, C);
}
